// RelationalLayerStackModule_21973052686563
// MI455X (gfx1250) — compile-verified, matrix operand order checked by reading, not run on hardware
//
#include <hip/hip_runtime.h>
#include <hip/hip_bf16.h>
#include <cstdint>
#include <cstddef>

// ---------------------------------------------------------------------------
// Types for CDNA5 WMMA (wave32): A/B = v16h (16 x f16), C/D = v8f (8 x f32)
// ---------------------------------------------------------------------------
typedef _Float16 half16 __attribute__((ext_vector_type(16)));
typedef _Float16 half8  __attribute__((ext_vector_type(8)));
typedef float    float8 __attribute__((ext_vector_type(8)));

#define NPW 4   // waves per block (128 threads)

static __device__ __forceinline__ float8 wmma16(half16 a, half16 b, float8 c) {
  // D = A(16x32 f16) * B(32x16 f16) + C(16x16 f32)
  return __builtin_amdgcn_wmma_f32_16x16x32_f16(false, a, false, b, (short)0, c,
                                                false, false);
}

static __device__ __forceinline__ half16 make_h16(half8 lo, half8 hi) {
  half16 r;
#pragma unroll
  for (int i = 0; i < 8; ++i) { r[i] = lo[i]; r[i + 8] = hi[i]; }
  return r;
}

// A-fragment load from an LDS tile laid out [16 rows][stride halves], row-major.
// ISA 16-bit A layout: lane m=l&15; lanes 0-15 hold K = {koff..koff+7, 16+koff..},
// lanes 16-31 the +8 K groups (koff = 8*(l>>4)).
static __device__ __forceinline__ half16 lds_afrag(const _Float16* base, int stride,
                                                   int kt, int lane) {
  const int m    = lane & 15;
  const int koff = (lane >> 4) << 3;
  const _Float16* row = base + m * stride + kt * 32;
  half8 lo = *(const half8*)(row + koff);
  half8 hi = *(const half8*)(row + 16 + koff);
  return make_h16(lo, hi);
}

// B-fragment load from pre-swizzled weights: fragment (kt,nt) stored as
// 32 lanes x 16 halves contiguous -> one coalesced 32B load per lane.
static __device__ __forceinline__ half16 gbl_bfrag(const _Float16* sw, int kt, int nt,
                                                   int ntn, int lane) {
  const _Float16* p = sw + ((((size_t)kt * ntn + nt) * 32 + lane) << 4);
  half8 lo = *(const half8*)p;
  half8 hi = *(const half8*)(p + 8);
  return make_h16(lo, hi);
}

// ---------------------------------------------------------------------------
// Weight pre-swizzle: f32 row-major [K][Nc] -> f16 B-fragment order.
// k_local(j,l) = 16*(j>>3) + (j&7) + 8*(l>>4); n = nt*16 + (l&15)
// ---------------------------------------------------------------------------
__global__ void swizzle_w(const float* __restrict__ src, _Float16* __restrict__ dst,
                          int K, int Nc) {
  int tid = blockIdx.x * blockDim.x + threadIdx.x;
  int total = K * Nc;
  if (tid >= total) return;
  int f = tid >> 9;          // 512 halves per fragment
  int rem = tid & 511;
  int l = rem >> 4;
  int j = rem & 15;
  int ntn = Nc >> 4;
  int kt = f / ntn, nt = f - kt * ntn;
  int k = kt * 32 + ((j >> 3) << 4) + (j & 7) + ((l >> 4) << 3);
  int n = nt * 16 + (l & 15);
  dst[tid] = (_Float16)src[k * Nc + n];
}

__global__ void zero_f32(float* p, size_t n) {
  size_t i = (size_t)blockIdx.x * blockDim.x + threadIdx.x;
  size_t s = (size_t)gridDim.x * blockDim.x;
  for (; i < n; i += s) p[i] = 0.f;
}
__global__ void zero_f16(_Float16* p, size_t n) {
  size_t i = (size_t)blockIdx.x * blockDim.x + threadIdx.x;
  size_t s = (size_t)gridDim.x * blockDim.x;
  for (; i < n; i += s) p[i] = (_Float16)0.f;
}

// ---------------------------------------------------------------------------
// Unary messages: msg = relu(h[idx] @ W1 + b1) @ W2 + b2 ; atomic scatter to agg
// One wave per 16-atom tile. K=64 (2 k-steps), N=64 (4 n-tiles).
// ---------------------------------------------------------------------------
__global__ __launch_bounds__(128) void unary_msg(
    const int* __restrict__ rel_unary, const _Float16* __restrict__ hf16,
    const _Float16* __restrict__ W1sw, const float* __restrict__ b1,
    const _Float16* __restrict__ W2sw, const float* __restrict__ b2,
    float* __restrict__ agg, int ntiles) {
  __shared__ __align__(16) _Float16 tile[NPW][16 * 64];
  __shared__ __align__(16) _Float16 mid[NPW][16 * 64];
  const int wave = threadIdx.x >> 5, lane = threadIdx.x & 31;
  const int t = blockIdx.x * NPW + wave;
  const bool active = (t < ntiles);
  const int tc = active ? t : (ntiles - 1);

  { // gather 16 rows x 64 halves: lane covers (row = lane&15, 32-half segment)
    const int m = lane & 15;
    const int seg = (lane >> 4) * 32;
    const int node = rel_unary[tc * 16 + m];
    const half8* sp = (const half8*)(hf16 + (size_t)node * 64 + seg);
    half8* dp = (half8*)(&tile[wave][m * 64 + seg]);
#pragma unroll
    for (int i = 0; i < 4; ++i) dp[i] = sp[i];
  }
  __syncthreads();

  const int m0 = (lane >> 4) << 3;
  const int nl = lane & 15;
#pragma unroll
  for (int nt = 0; nt < 4; ++nt) {
    float8 acc = {0.f, 0.f, 0.f, 0.f, 0.f, 0.f, 0.f, 0.f};
#pragma unroll
    for (int kt = 0; kt < 2; ++kt)
      acc = wmma16(lds_afrag(&tile[wave][0], 64, kt, lane),
                   gbl_bfrag(W1sw, kt, nt, 4, lane), acc);
    const float bias = b1[nt * 16 + nl];
    const int n = nt * 16 + nl;
#pragma unroll
    for (int e = 0; e < 8; ++e) {
      float v = acc[e] + bias;
      mid[wave][(m0 + e) * 64 + n] = (_Float16)(v > 0.f ? v : 0.f);
    }
  }
  __syncthreads();

#pragma unroll
  for (int nt = 0; nt < 4; ++nt) {
    float8 acc = {0.f, 0.f, 0.f, 0.f, 0.f, 0.f, 0.f, 0.f};
#pragma unroll
    for (int kt = 0; kt < 2; ++kt)
      acc = wmma16(lds_afrag(&mid[wave][0], 64, kt, lane),
                   gbl_bfrag(W2sw, kt, nt, 4, lane), acc);
    const float bias = b2[nt * 16 + nl];
    if (active) {
      const int n = nt * 16 + nl;
#pragma unroll
      for (int e = 0; e < 8; ++e) {
        const int node = rel_unary[tc * 16 + m0 + e];
        atomicAdd(&agg[(size_t)node * 64 + n], acc[e] + bias);
      }
    }
  }
}

// ---------------------------------------------------------------------------
// Edge messages: xe = [h[a0], h[a1]] (16x128); MLP 128->128->128;
// cols 0..63 scatter to a0, cols 64..127 scatter to a1.
// ---------------------------------------------------------------------------
__global__ __launch_bounds__(128) void edge_msg(
    const int* __restrict__ rel_edge, const _Float16* __restrict__ hf16,
    const _Float16* __restrict__ W1sw, const float* __restrict__ b1,
    const _Float16* __restrict__ W2sw, const float* __restrict__ b2,
    float* __restrict__ agg, int ntiles) {
  __shared__ __align__(16) _Float16 tile[NPW][16 * 128];
  __shared__ __align__(16) _Float16 mid[NPW][16 * 128];
  const int wave = threadIdx.x >> 5, lane = threadIdx.x & 31;
  const int t = blockIdx.x * NPW + wave;
  const bool active = (t < ntiles);
  const int tc = active ? t : (ntiles - 1);

  { // gather: lane -> (edge m = lane&15, arg p = lane>>4), 64 halves each
    const int m = lane & 15, p = lane >> 4;
    const int node = rel_edge[(tc * 16 + m) * 2 + p];
    const half8* sp = (const half8*)(hf16 + (size_t)node * 64);
    half8* dp = (half8*)(&tile[wave][m * 128 + p * 64]);
#pragma unroll
    for (int i = 0; i < 8; ++i) dp[i] = sp[i];
  }
  __syncthreads();

  const int m0 = (lane >> 4) << 3;
  const int nl = lane & 15;

  half16 afr[4];
#pragma unroll
  for (int kt = 0; kt < 4; ++kt) afr[kt] = lds_afrag(&tile[wave][0], 128, kt, lane);

#pragma unroll
  for (int nt = 0; nt < 8; ++nt) {
    float8 acc = {0.f, 0.f, 0.f, 0.f, 0.f, 0.f, 0.f, 0.f};
#pragma unroll
    for (int kt = 0; kt < 4; ++kt)
      acc = wmma16(afr[kt], gbl_bfrag(W1sw, kt, nt, 8, lane), acc);
    const float bias = b1[nt * 16 + nl];
    const int n = nt * 16 + nl;
#pragma unroll
    for (int e = 0; e < 8; ++e) {
      float v = acc[e] + bias;
      mid[wave][(m0 + e) * 128 + n] = (_Float16)(v > 0.f ? v : 0.f);
    }
  }
  __syncthreads();

  half16 afr2[4];
#pragma unroll
  for (int kt = 0; kt < 4; ++kt) afr2[kt] = lds_afrag(&mid[wave][0], 128, kt, lane);

#pragma unroll
  for (int nt = 0; nt < 8; ++nt) {
    float8 acc = {0.f, 0.f, 0.f, 0.f, 0.f, 0.f, 0.f, 0.f};
#pragma unroll
    for (int kt = 0; kt < 4; ++kt)
      acc = wmma16(afr2[kt], gbl_bfrag(W2sw, kt, nt, 8, lane), acc);
    const float bias = b2[nt * 16 + nl];
    if (active) {
      const int arg = (nt >= 4) ? 1 : 0;
      const int col = (nt - 4 * arg) * 16 + nl;   // column inside 64-wide message
#pragma unroll
      for (int e = 0; e < 8; ++e) {
        const int node = rel_edge[(tc * 16 + m0 + e) * 2 + arg];
        atomicAdd(&agg[(size_t)node * 64 + col], acc[e] + bias);
      }
    }
  }
}

// ---------------------------------------------------------------------------
// Node update: nxt = LN( relu([h,agg] @ Wu1 + bu1) @ Wu2 + bu2 )
// ---------------------------------------------------------------------------
__global__ __launch_bounds__(128) void node_update(
    const _Float16* __restrict__ hf16, const float* __restrict__ agg,
    const _Float16* __restrict__ W1sw, const float* __restrict__ b1,
    const _Float16* __restrict__ W2sw, const float* __restrict__ b2,
    const float* __restrict__ ln_w, const float* __restrict__ ln_b,
    float* __restrict__ nxt, int ntiles) {
  __shared__ __align__(16) _Float16 tile[NPW][16 * 128];
  __shared__ __align__(16) _Float16 mid[NPW][16 * 64];
  __shared__ __align__(16) float    outt[NPW][16 * 64];
  const int wave = threadIdx.x >> 5, lane = threadIdx.x & 31;
  const int t = blockIdx.x * NPW + wave;
  const bool active = (t < ntiles);
  const int tc = active ? t : (ntiles - 1);

  { // build input tile: [h | f16(agg)]
    const int m = lane & 15, p = lane >> 4;
    const int node = tc * 16 + m;
    if (p == 0) {
      const half8* sp = (const half8*)(hf16 + (size_t)node * 64);
      half8* dp = (half8*)(&tile[wave][m * 128]);
#pragma unroll
      for (int i = 0; i < 8; ++i) dp[i] = sp[i];
    } else {
      const float* ar = agg + (size_t)node * 64;
      _Float16* dr = &tile[wave][m * 128 + 64];
#pragma unroll
      for (int i = 0; i < 64; ++i) dr[i] = (_Float16)ar[i];
    }
  }
  __syncthreads();

  const int m0 = (lane >> 4) << 3;
  const int nl = lane & 15;

  half16 afr[4];
#pragma unroll
  for (int kt = 0; kt < 4; ++kt) afr[kt] = lds_afrag(&tile[wave][0], 128, kt, lane);

#pragma unroll
  for (int nt = 0; nt < 4; ++nt) {
    float8 acc = {0.f, 0.f, 0.f, 0.f, 0.f, 0.f, 0.f, 0.f};
#pragma unroll
    for (int kt = 0; kt < 4; ++kt)
      acc = wmma16(afr[kt], gbl_bfrag(W1sw, kt, nt, 4, lane), acc);
    const float bias = b1[nt * 16 + nl];
    const int n = nt * 16 + nl;
#pragma unroll
    for (int e = 0; e < 8; ++e) {
      float v = acc[e] + bias;
      mid[wave][(m0 + e) * 64 + n] = (_Float16)(v > 0.f ? v : 0.f);
    }
  }
  __syncthreads();

#pragma unroll
  for (int nt = 0; nt < 4; ++nt) {
    float8 acc = {0.f, 0.f, 0.f, 0.f, 0.f, 0.f, 0.f, 0.f};
#pragma unroll
    for (int kt = 0; kt < 2; ++kt)
      acc = wmma16(lds_afrag(&mid[wave][0], 64, kt, lane),
                   gbl_bfrag(W2sw, kt, nt, 4, lane), acc);
    const float bias = b2[nt * 16 + nl];
    const int n = nt * 16 + nl;
#pragma unroll
    for (int e = 0; e < 8; ++e) outt[wave][(m0 + e) * 64 + n] = acc[e] + bias;
  }
  __syncthreads();

  if (lane < 16) { // per-row LayerNorm
    const float* row = &outt[wave][lane * 64];
    float mu = 0.f;
#pragma unroll
    for (int i = 0; i < 64; ++i) mu += row[i];
    mu *= (1.f / 64.f);
    float var = 0.f;
#pragma unroll
    for (int i = 0; i < 64; ++i) { float d = row[i] - mu; var += d * d; }
    var *= (1.f / 64.f);
    const float inv = rsqrtf(var + 1e-5f);
    if (active) {
      const size_t node = (size_t)(tc * 16 + lane);
#pragma unroll
      for (int i = 0; i < 64; ++i)
        nxt[node * 64 + i] = (row[i] - mu) * inv * ln_w[i] + ln_b[i];
    }
  }
}

// ---------------------------------------------------------------------------
// Graph readout: per-graph sum of h (contiguous graph_ids) then @ Wr + br
// ---------------------------------------------------------------------------
__global__ void graph_sum(const float* __restrict__ h, float* __restrict__ gpre,
                          int npg) {
  const int g = blockIdx.x;
  const int col = threadIdx.x & 63;
  const int rg = threadIdx.x >> 6;    // 0..3
  float s = 0.f;
  for (int r = rg; r < npg; r += 4)
    s += h[((size_t)g * npg + r) * 64 + col];
  __shared__ float red[4][64];
  red[rg][col] = s;
  __syncthreads();
  if (rg == 0)
    gpre[g * 64 + col] = red[0][col] + red[1][col] + red[2][col] + red[3][col];
}

__global__ void graph_proj(const float* __restrict__ gpre, const float* __restrict__ Wr,
                           const float* __restrict__ br, float* __restrict__ g,
                           int total) {
  int tid = blockIdx.x * blockDim.x + threadIdx.x;
  if (tid >= total) return;
  int gi = tid >> 6, n = tid & 63;
  float s = br[n];
#pragma unroll
  for (int k = 0; k < 64; ++k) s += gpre[gi * 64 + k] * Wr[k * 64 + n];
  g[tid] = s;
}

// ---------------------------------------------------------------------------
// Global update + residual: gm = LN(MLP([h, g[gid]])); h' = h + nxt + gm
// ---------------------------------------------------------------------------
__global__ __launch_bounds__(128) void global_update(
    const _Float16* __restrict__ hf16, const float* __restrict__ hf32,
    const float* __restrict__ nxt, const float* __restrict__ gmat,
    const int* __restrict__ graph_ids,
    const _Float16* __restrict__ W1sw, const float* __restrict__ b1,
    const _Float16* __restrict__ W2sw, const float* __restrict__ b2,
    const float* __restrict__ ln_w, const float* __restrict__ ln_b,
    float* __restrict__ hout32, _Float16* __restrict__ hout16, int ntiles) {
  __shared__ __align__(16) _Float16 tile[NPW][16 * 128];
  __shared__ __align__(16) _Float16 mid[NPW][16 * 64];
  __shared__ __align__(16) float    outt[NPW][16 * 64];
  const int wave = threadIdx.x >> 5, lane = threadIdx.x & 31;
  const int t = blockIdx.x * NPW + wave;
  const bool active = (t < ntiles);
  const int tc = active ? t : (ntiles - 1);

  {
    const int m = lane & 15, p = lane >> 4;
    const int node = tc * 16 + m;
    if (p == 0) {
      const half8* sp = (const half8*)(hf16 + (size_t)node * 64);
      half8* dp = (half8*)(&tile[wave][m * 128]);
#pragma unroll
      for (int i = 0; i < 8; ++i) dp[i] = sp[i];
    } else {
      const int gi = graph_ids[node];
      const float* gr = gmat + (size_t)gi * 64;
      _Float16* dr = &tile[wave][m * 128 + 64];
#pragma unroll
      for (int i = 0; i < 64; ++i) dr[i] = (_Float16)gr[i];
    }
  }
  __syncthreads();

  const int m0 = (lane >> 4) << 3;
  const int nl = lane & 15;

  half16 afr[4];
#pragma unroll
  for (int kt = 0; kt < 4; ++kt) afr[kt] = lds_afrag(&tile[wave][0], 128, kt, lane);

#pragma unroll
  for (int nt = 0; nt < 4; ++nt) {
    float8 acc = {0.f, 0.f, 0.f, 0.f, 0.f, 0.f, 0.f, 0.f};
#pragma unroll
    for (int kt = 0; kt < 4; ++kt)
      acc = wmma16(afr[kt], gbl_bfrag(W1sw, kt, nt, 4, lane), acc);
    const float bias = b1[nt * 16 + nl];
    const int n = nt * 16 + nl;
#pragma unroll
    for (int e = 0; e < 8; ++e) {
      float v = acc[e] + bias;
      mid[wave][(m0 + e) * 64 + n] = (_Float16)(v > 0.f ? v : 0.f);
    }
  }
  __syncthreads();

#pragma unroll
  for (int nt = 0; nt < 4; ++nt) {
    float8 acc = {0.f, 0.f, 0.f, 0.f, 0.f, 0.f, 0.f, 0.f};
#pragma unroll
    for (int kt = 0; kt < 2; ++kt)
      acc = wmma16(lds_afrag(&mid[wave][0], 64, kt, lane),
                   gbl_bfrag(W2sw, kt, nt, 4, lane), acc);
    const float bias = b2[nt * 16 + nl];
    const int n = nt * 16 + nl;
#pragma unroll
    for (int e = 0; e < 8; ++e) outt[wave][(m0 + e) * 64 + n] = acc[e] + bias;
  }
  __syncthreads();

  if (lane < 16) {
    const float* row = &outt[wave][lane * 64];
    float mu = 0.f;
#pragma unroll
    for (int i = 0; i < 64; ++i) mu += row[i];
    mu *= (1.f / 64.f);
    float var = 0.f;
#pragma unroll
    for (int i = 0; i < 64; ++i) { float d = row[i] - mu; var += d * d; }
    var *= (1.f / 64.f);
    const float inv = rsqrtf(var + 1e-5f);
    if (active) {
      const size_t node = (size_t)(tc * 16 + lane);
#pragma unroll
      for (int i = 0; i < 64; ++i) {
        float gm = (row[i] - mu) * inv * ln_w[i] + ln_b[i];
        float hn = hf32[node * 64 + i] + nxt[node * 64 + i] + gm;
        hout32[node * 64 + i] = hn;
        hout16[node * 64 + i] = (_Float16)hn;
      }
    }
  }
}

// ---------------------------------------------------------------------------
// Host orchestration
// ---------------------------------------------------------------------------
extern "C" void kernel_launch(void* const* d_in, const int* in_sizes, int n_in,
                              void* d_out, int out_size, void* d_ws, size_t ws_size,
                              hipStream_t stream) {
  const int* rel_unary = (const int*)d_in[0];
  const int* rel_edge  = (const int*)d_in[1];
  const int* graph_ids = (const int*)d_in[2];
  const float* Wm1_u = (const float*)d_in[4];  const float* bm1_u = (const float*)d_in[5];
  const float* Wm2_u = (const float*)d_in[6];  const float* bm2_u = (const float*)d_in[7];
  const float* Wm1_e = (const float*)d_in[8];  const float* bm1_e = (const float*)d_in[9];
  const float* Wm2_e = (const float*)d_in[10]; const float* bm2_e = (const float*)d_in[11];
  const float* Wu1   = (const float*)d_in[12]; const float* bu1   = (const float*)d_in[13];
  const float* Wu2   = (const float*)d_in[14]; const float* bu2   = (const float*)d_in[15];
  const float* ln_w  = (const float*)d_in[16]; const float* ln_b  = (const float*)d_in[17];
  const float* Wr    = (const float*)d_in[18]; const float* br    = (const float*)d_in[19];
  const float* Wg1   = (const float*)d_in[20]; const float* bg1   = (const float*)d_in[21];
  const float* Wg2   = (const float*)d_in[22]; const float* bg2   = (const float*)d_in[23];

  const int MU = in_sizes[0];          // 100000
  const int ME = in_sizes[1] / 2;      // 200000
  const int N  = in_sizes[2];          // 50000
  const int G  = 100;                  // reference constant (scalar lives on device)
  const int npg = N / G;
  const int L  = 4;

  // workspace carve-up
  char* w = (char*)d_ws;
  auto carve = [&](size_t bytes) {
    void* p = (void*)w;
    w += (bytes + 255) & ~(size_t)255;
    return p;
  };
  float*     h32  = (float*)carve((size_t)N * 64 * 4);
  _Float16*  h16  = (_Float16*)carve((size_t)N * 64 * 2);
  float*     agg  = (float*)carve((size_t)N * 64 * 4);
  float*     nxt  = (float*)carve((size_t)N * 64 * 4);
  float*     gpre = (float*)carve((size_t)G * 64 * 4);
  float*     gmat = (float*)carve((size_t)G * 64 * 4);
  _Float16* sWm1u = (_Float16*)carve(64 * 64 * 2);
  _Float16* sWm2u = (_Float16*)carve(64 * 64 * 2);
  _Float16* sWm1e = (_Float16*)carve(128 * 128 * 2);
  _Float16* sWm2e = (_Float16*)carve(128 * 128 * 2);
  _Float16* sWu1  = (_Float16*)carve(128 * 64 * 2);
  _Float16* sWu2  = (_Float16*)carve(64 * 64 * 2);
  _Float16* sWg1  = (_Float16*)carve(128 * 64 * 2);
  _Float16* sWg2  = (_Float16*)carve(64 * 64 * 2);

  auto cdiv = [](int a, int b) { return (a + b - 1) / b; };

  // h starts at zero (and d_ws is poisoned, so initialize)
  zero_f32<<<512, 256, 0, stream>>>(h32, (size_t)N * 64);
  zero_f16<<<512, 256, 0, stream>>>(h16, (size_t)N * 64);

  // weight pre-swizzle into WMMA B-fragment layout
  swizzle_w<<<cdiv(64 * 64, 256), 256, 0, stream>>>(Wm1_u, sWm1u, 64, 64);
  swizzle_w<<<cdiv(64 * 64, 256), 256, 0, stream>>>(Wm2_u, sWm2u, 64, 64);
  swizzle_w<<<cdiv(128 * 128, 256), 256, 0, stream>>>(Wm1_e, sWm1e, 128, 128);
  swizzle_w<<<cdiv(128 * 128, 256), 256, 0, stream>>>(Wm2_e, sWm2e, 128, 128);
  swizzle_w<<<cdiv(128 * 64, 256), 256, 0, stream>>>(Wu1, sWu1, 128, 64);
  swizzle_w<<<cdiv(64 * 64, 256), 256, 0, stream>>>(Wu2, sWu2, 64, 64);
  swizzle_w<<<cdiv(128 * 64, 256), 256, 0, stream>>>(Wg1, sWg1, 128, 64);
  swizzle_w<<<cdiv(64 * 64, 256), 256, 0, stream>>>(Wg2, sWg2, 64, 64);

  const int tU = cdiv(MU, 16), tE = cdiv(ME, 16), tN = cdiv(N, 16);

  for (int layer = 0; layer < L; ++layer) {
    zero_f32<<<512, 256, 0, stream>>>(agg, (size_t)N * 64);

    unary_msg<<<cdiv(tU, NPW), 128, 0, stream>>>(
        rel_unary, h16, sWm1u, bm1_u, sWm2u, bm2_u, agg, tU);

    edge_msg<<<cdiv(tE, NPW), 128, 0, stream>>>(
        rel_edge, h16, sWm1e, bm1_e, sWm2e, bm2_e, agg, tE);

    graph_sum<<<G, 256, 0, stream>>>(h32, gpre, npg);
    graph_proj<<<cdiv(G * 64, 256), 256, 0, stream>>>(gpre, Wr, br, gmat, G * 64);

    node_update<<<cdiv(tN, NPW), 128, 0, stream>>>(
        h16, agg, sWu1, bu1, sWu2, bu2, ln_w, ln_b, nxt, tN);

    float* out32 = (layer == L - 1) ? (float*)d_out : h32;
    global_update<<<cdiv(tN, NPW), 128, 0, stream>>>(
        h16, h32, nxt, gmat, graph_ids, sWg1, bg1, sWg2, bg2, ln_w, ln_b,
        out32, h16, tN);
  }
}
